// TransformerModel_38620345926380
// MI455X (gfx1250) — matrix operand order repaired and checked by reading, not compiled, not run on hardware
//
#include <hip/hip_runtime.h>

typedef __attribute__((ext_vector_type(16))) _Float16 v16h;
typedef __attribute__((ext_vector_type(8)))  _Float16 v8h;
typedef __attribute__((ext_vector_type(8)))  float    v8f;

#define VOCAB  65
#define DMODEL 384
#define CTX    256
#define NHEAD  64
#define HSZ    6
#define NLAYER 6
#define BATCH  32
#define MROWS  (BATCH*CTX)   /* 8192 token rows */
#define DFF    (4*DMODEL)    /* 1536 */
#define DQKV   (3*DMODEL)    /* 1152: fused q|k|v columns */
#define LN_EPS 1e-5f

// ---------------------------------------------------------------------------
// Embedding: x[row,d] = tok_emb[idx[row],d] + pos_emb[row%T, d]
// ---------------------------------------------------------------------------
__global__ void embed_kernel(const int* __restrict__ idx,
                             const float* __restrict__ tok,
                             const float* __restrict__ pos,
                             float* __restrict__ x) {
  int i = blockIdx.x * blockDim.x + threadIdx.x;
  if (i >= MROWS * DMODEL) return;
  int row = i / DMODEL, d = i % DMODEL;
  int t = row % CTX;
  x[i] = tok[idx[row] * DMODEL + d] + pos[t * DMODEL + d];
}

// ---------------------------------------------------------------------------
// Weight convert + transpose: f32 [K,N] -> f16 [N,K] (row stride K)
// Coalesced reads; the scatter happens once per layer instead of per GEMM tile.
// ---------------------------------------------------------------------------
__global__ void cvt_t_kernel(const float* __restrict__ s, _Float16* __restrict__ d,
                             int K, int N) {
  int i = blockIdx.x * blockDim.x + threadIdx.x;
  if (i >= K * N) return;
  int k = i / N, n = i % N;
  d[(size_t)n * K + k] = (_Float16)s[i];
}

// ---------------------------------------------------------------------------
// Row LayerNorm over D=384, f32 in -> f16 out (GEMM operand)
// ---------------------------------------------------------------------------
__global__ __launch_bounds__(128)
void ln_kernel(const float* __restrict__ x,
               const float* __restrict__ g,
               const float* __restrict__ b,
               _Float16* __restrict__ out) {
  __shared__ float r1[128], r2[128];
  int row = blockIdx.x, tid = threadIdx.x;
  const float* xr = x + (size_t)row * DMODEL;
  float s = 0.f, sq = 0.f;
  for (int d = tid; d < DMODEL; d += 128) { float v = xr[d]; s += v; sq += v * v; }
  r1[tid] = s; r2[tid] = sq; __syncthreads();
  for (int off = 64; off > 0; off >>= 1) {
    if (tid < off) { r1[tid] += r1[tid + off]; r2[tid] += r2[tid + off]; }
    __syncthreads();
  }
  float mean = r1[0] * (1.f / DMODEL);
  float var  = r2[0] * (1.f / DMODEL) - mean * mean;
  float rstd = rsqrtf(var + LN_EPS);
  _Float16* o = out + (size_t)row * DMODEL;
  for (int d = tid; d < DMODEL; d += 128)
    o[d] = (_Float16)((xr[d] - mean) * rstd * g[d] + b[d]);
}

// ---------------------------------------------------------------------------
// Tiled WMMA GEMM:  C[M,N] = A[M,K](f16) @ Bt[N,K](f16, pre-transposed)
// Block: 128 threads = 4 waves; tile 128(M) x 64(N); K panel = 32.
// Each wave owns a 32x64 strip -> 8 x v_wmma_f32_16x16x32_f16 per panel,
// with each B fragment reused across two M-subtiles.
// Fragment layouts per CDNA5 ISA 7.12.2:
//   A 16x32 f16: lane m=lane&15, kh=lane>>4; elem e: k = (e>>3)*16 + kh*8 + (e&7)
//   B 32x16 f16: lane n=lane&15, kh=lane>>4; elem e: k = (e>>3)*16 + kh*8 + (e&7)
//   C/D 16x16 f32: vgpr r: m = (lane>>4)*8 + r, n = lane&15
// Both panels stage as contiguous 16-byte chunks (A row-major, Bt row-major).
// ---------------------------------------------------------------------------
__global__ __launch_bounds__(128)
void gemm_wmma(const _Float16* __restrict__ A,   // [M,K]
               const _Float16* __restrict__ Bt,  // [N,K]
               const float* __restrict__ bias,   // [N] or null
               const float* __restrict__ resid,  // [M,N] or null
               float* __restrict__ outF32,       // [M,N] or null
               _Float16* __restrict__ outF16,    // [M,N] or null
               int M, int N, int K, int relu) {
  __shared__ _Float16 As[128][40];  // [m][k], 40-half pitch (80B, 16B aligned)
  __shared__ _Float16 Bs[64][40];   // [n][k]
  const int tid   = threadIdx.x;
  const int lane  = tid & 31;
  const int wave  = tid >> 5;
  const int tileM = blockIdx.x * 128;
  const int tileN = blockIdx.y * 64;

  // staging geometry: thread owns row (tid>>2) (+32*j) and 8-half chunk (tid&3)*8
  const int    srow = tid >> 2;
  const int    scol = (tid & 3) * 8;
  const size_t rstep = (size_t)32 * K;  // 32 rows
  const _Float16* aBase = A  + (size_t)(tileM + srow) * K + scol;
  const _Float16* bBase = Bt + (size_t)(tileN + srow) * K + scol;

  v8f acc[2][4] = {};

  const int mh = lane & 15;
  const int kh = lane >> 4;

  for (int k0 = 0; k0 < K; k0 += 32) {
    __syncthreads();
    // stage A panel 128x32: 4 contiguous 16B chunks per thread
#pragma unroll
    for (int j = 0; j < 4; ++j)
      *(v8h*)&As[srow + 32 * j][scol] = *(const v8h*)(aBase + k0 + rstep * j);
    // stage B panel 64x32: 2 chunks per thread, zero-fill beyond N (head N=65)
#pragma unroll
    for (int j = 0; j < 2; ++j) {
      v8h bv = {};
      if (tileN + srow + 32 * j < N) bv = *(const v8h*)(bBase + k0 + rstep * j);
      *(v8h*)&Bs[srow + 32 * j][scol] = bv;
    }
    // prefetch next K panel while this one is consumed (global_prefetch_b8)
    if (k0 + 32 < K) {
      __builtin_prefetch(aBase + k0 + 32, 0, 0);
      __builtin_prefetch(bBase + k0 + 32, 0, 0);
    }
    __syncthreads();

    v16h af[2];
#pragma unroll
    for (int mt = 0; mt < 2; ++mt) {
      const int m = wave * 32 + mt * 16 + mh;
#pragma unroll
      for (int j = 0; j < 8; ++j) af[mt][j]     = As[m][kh * 8 + j];
#pragma unroll
      for (int j = 0; j < 8; ++j) af[mt][8 + j] = As[m][16 + kh * 8 + j];
    }
#pragma unroll
    for (int nt = 0; nt < 4; ++nt) {
      v16h bf;
      const int n = nt * 16 + mh;
#pragma unroll
      for (int j = 0; j < 8; ++j) bf[j]     = Bs[n][kh * 8 + j];
#pragma unroll
      for (int j = 0; j < 8; ++j) bf[8 + j] = Bs[n][16 + kh * 8 + j];
#pragma unroll
      for (int mt = 0; mt < 2; ++mt)
        acc[mt][nt] = __builtin_amdgcn_wmma_f32_16x16x32_f16(
            false, af[mt], false, bf, (short)0, acc[mt][nt], false, false);
    }
  }

  // epilogue: bias / relu / residual, f32 and/or f16 stores
#pragma unroll
  for (int mt = 0; mt < 2; ++mt) {
#pragma unroll
    for (int r = 0; r < 8; ++r) {
      const int gm = tileM + wave * 32 + mt * 16 + (lane >> 4) * 8 + r;
#pragma unroll
      for (int nt = 0; nt < 4; ++nt) {
        const int gn = tileN + nt * 16 + mh;
        if (gn < N) {
          float v = acc[mt][nt][r];
          if (bias)  v += bias[gn];
          if (relu)  v  = fmaxf(v, 0.f);
          if (resid) v += resid[(size_t)gm * N + gn];
          if (outF32) outF32[(size_t)gm * N + gn] = v;
          if (outF16) outF16[(size_t)gm * N + gn] = (_Float16)v;
        }
      }
    }
  }
}

// ---------------------------------------------------------------------------
// Causal attention, one workgroup per (batch, head). HS=6 (too small for
// WMMA K) -> exact online-softmax (flash) per query row.
// qkv is the fused f16 [MROWS, 1152] tensor: q @ col h*6, k @ 384+h*6,
// v @ 768+h*6. Output o is f16 [MROWS, 384].
// ---------------------------------------------------------------------------
__global__ __launch_bounds__(256)
void attn_kernel(const _Float16* __restrict__ qkv,
                 _Float16* __restrict__ o) {
  __shared__ float kf[CTX][8];
  __shared__ float vf[CTX][8];
  const int b = blockIdx.x / NHEAD;
  const int h = blockIdx.x % NHEAD;
  const int t = threadIdx.x;
  const int row = b * CTX + t;
  const size_t qoff = (size_t)row * DQKV + h * HSZ;

  float qv[HSZ];
#pragma unroll
  for (int d = 0; d < HSZ; ++d) {
    qv[d]    = (float)qkv[qoff + d];
    kf[t][d] = (float)qkv[qoff + DMODEL + d];
    vf[t][d] = (float)qkv[qoff + 2 * DMODEL + d];
  }
  __syncthreads();

  const float scale = 0.40824829046386302f;  // 1/sqrt(6)
  float mx = -3.0e38f, l = 0.f, acc[HSZ];
#pragma unroll
  for (int d = 0; d < HSZ; ++d) acc[d] = 0.f;

  for (int j = 0; j <= t; ++j) {  // causal: j <= t
    float s = 0.f;
#pragma unroll
    for (int d = 0; d < HSZ; ++d) s += qv[d] * kf[j][d];
    s *= scale;
    float nm = fmaxf(mx, s);
    float cf = __expf(mx - nm);
    float p  = __expf(s - nm);
    l = l * cf + p;
#pragma unroll
    for (int d = 0; d < HSZ; ++d) acc[d] = acc[d] * cf + p * vf[j][d];
    mx = nm;
  }
  const float inv = 1.f / l;
  _Float16* orow = o + (size_t)row * DMODEL + h * HSZ;
#pragma unroll
  for (int d = 0; d < HSZ; ++d) orow[d] = (_Float16)(acc[d] * inv);
}

// ---------------------------------------------------------------------------
// Host-side orchestration (graph-capture safe: stream-only, no alloc/sync)
// ---------------------------------------------------------------------------
extern "C" void kernel_launch(void* const* d_in, const int* in_sizes, int n_in,
                              void* d_out, int out_size, void* d_ws, size_t ws_size,
                              hipStream_t stream) {
  (void)in_sizes; (void)n_in; (void)out_size; (void)ws_size;
  const int*   idx  = (const int*)  d_in[0];
  const float* tok  = (const float*)d_in[1];
  const float* pos  = (const float*)d_in[2];
  const float* wq   = (const float*)d_in[3];
  const float* wk   = (const float*)d_in[4];
  const float* wv   = (const float*)d_in[5];
  const float* wo   = (const float*)d_in[6];
  const float* bo   = (const float*)d_in[7];
  const float* ln1g = (const float*)d_in[8];
  const float* ln1b = (const float*)d_in[9];
  const float* ln2g = (const float*)d_in[10];
  const float* ln2b = (const float*)d_in[11];
  const float* w1   = (const float*)d_in[12];
  const float* b1   = (const float*)d_in[13];
  const float* w2   = (const float*)d_in[14];
  const float* b2   = (const float*)d_in[15];
  const float* lnfg = (const float*)d_in[16];
  const float* lnfb = (const float*)d_in[17];
  const float* wh   = (const float*)d_in[18];
  const float* bh   = (const float*)d_in[19];
  float* out = (float*)d_out;

  // workspace carve-out (~73 MB total)
  char* ws = (char*)d_ws;
  size_t cur = 0;
  auto carve = [&](size_t bytes) -> void* {
    void* p = ws + cur;
    cur += (bytes + 255) & ~(size_t)255;
    return p;
  };
  float*    x      = (float*)   carve((size_t)MROWS * DMODEL * 4);
  _Float16* h16    = (_Float16*)carve((size_t)MROWS * DMODEL * 2);
  _Float16* qkv16  = (_Float16*)carve((size_t)MROWS * DQKV * 2);
  _Float16* o16    = (_Float16*)carve((size_t)MROWS * DMODEL * 2);
  _Float16* f16b   = (_Float16*)carve((size_t)MROWS * DFF * 2);
  _Float16* wqkvT  = (_Float16*)carve((size_t)DQKV * DMODEL * 2);  // [1152,384]
  _Float16* woT    = (_Float16*)carve((size_t)DMODEL * DMODEL * 2);
  _Float16* w1T    = (_Float16*)carve((size_t)DFF * DMODEL * 2);   // [1536,384]
  _Float16* w2T    = (_Float16*)carve((size_t)DMODEL * DFF * 2);   // [384,1536]
  _Float16* whT    = (_Float16*)carve((size_t)VOCAB * DMODEL * 2); // [65,384]

  auto cvtT = [&](const float* s, _Float16* d, int K, int N) {
    cvt_t_kernel<<<(K * N + 255) / 256, 256, 0, stream>>>(s, d, K, N);
  };
  auto gemm = [&](const _Float16* A, const _Float16* Bm, const float* bias,
                  const float* resid, float* oF32, _Float16* oF16,
                  int N, int K, int relu) {
    dim3 g(MROWS / 128, (N + 63) / 64);
    gemm_wmma<<<g, 128, 0, stream>>>(A, Bm, bias, resid, oF32, oF16,
                                     MROWS, N, K, relu);
  };

  // embedding
  embed_kernel<<<(MROWS * DMODEL + 255) / 256, 256, 0, stream>>>(idx, tok, pos, x);

  const size_t DD = (size_t)DMODEL * DMODEL;
  const size_t DF = (size_t)DMODEL * DFF;

  for (int l = 0; l < NLAYER; ++l) {
    // per-layer weight convert+transpose into reused f16 slabs
    cvtT(wq + l * DD, wqkvT,                          DMODEL, DMODEL); // rows 0..383
    cvtT(wk + l * DD, wqkvT + (size_t)DMODEL * DMODEL, DMODEL, DMODEL); // rows 384..767
    cvtT(wv + l * DD, wqkvT + (size_t)2 * DMODEL * DMODEL, DMODEL, DMODEL);
    cvtT(wo + l * DD, woT, DMODEL, DMODEL);
    cvtT(w1 + l * DF, w1T, DMODEL, DFF);
    cvtT(w2 + l * DF, w2T, DFF, DMODEL);

    // h = LN1(x)
    ln_kernel<<<MROWS, 128, 0, stream>>>(x, ln1g + l * DMODEL, ln1b + l * DMODEL, h16);

    // fused qkv = h @ [Wq|Wk|Wv]  (no bias in reference)
    gemm(h16, wqkvT, nullptr, nullptr, nullptr, qkv16, DQKV, DMODEL, 0);

    // causal attention per (b, head)
    attn_kernel<<<BATCH * NHEAD, CTX, 0, stream>>>(qkv16, o16);

    // x = x + o @ Wo + Bo
    gemm(o16, woT, bo + l * DMODEL, x, x, nullptr, DMODEL, DMODEL, 0);

    // h2 = LN2(x)
    ln_kernel<<<MROWS, 128, 0, stream>>>(x, ln2g + l * DMODEL, ln2b + l * DMODEL, h16);

    // ffn: x = x + relu(h2 @ W1 + B1) @ W2 + B2
    gemm(h16, w1T, b1 + l * DFF, nullptr, nullptr, f16b, DFF, DMODEL, 1);
    gemm(f16b, w2T, b2 + l * DMODEL, x, x, nullptr, DMODEL, DFF, 0);
  }

  // final LN + head
  ln_kernel<<<MROWS, 128, 0, stream>>>(x, lnfg, lnfb, h16);
  cvtT(wh, whT, DMODEL, VOCAB);
  gemm(h16, whT, bh, nullptr, out, nullptr, VOCAB, DMODEL, 0);
}
